// ModelXLSTM_39711267619398
// MI455X (gfx1250) — matrix operand order repaired and checked by reading, not compiled, not run on hardware
//
#include <hip/hip_runtime.h>
#include <hip/hip_bf16.h>
#include <math.h>
#include <stdint.h>

// ---------------------------------------------------------------------------
// CDNA5 (gfx1250) wave32 WMMA types
// ---------------------------------------------------------------------------
typedef __attribute__((ext_vector_type(16))) __bf16 v16bf;
typedef __attribute__((ext_vector_type(8)))  float  v8f;
typedef unsigned short ushort_t;

union V16 {              // two b128 loads == one 16x32 bf16 fragment
  uint4 u4[2];
  v16bf v;
};

__device__ __forceinline__ ushort_t f2bfu(float f) {   // RNE f32 -> bf16 bits
  union { float f; unsigned u; } a; a.f = f;
  unsigned u = a.u;
  u += 0x7FFFu + ((u >> 16) & 1u);
  return (ushort_t)(u >> 16);
}
__device__ __forceinline__ float bf2f(ushort_t s) {
  union { unsigned u; float f; } a; a.u = ((unsigned)s) << 16;
  return a.f;
}
__device__ __forceinline__ __bf16 bfh(ushort_t s) {
  union { ushort_t s; __bf16 b; } o; o.s = s;
  return o.b;
}
__device__ __forceinline__ float lsig(float x) {          // log(sigmoid(x))
  return fminf(x, 0.0f) - log1pf(__expf(-fabsf(x)));
}
__device__ __forceinline__ float siluf(float x) { return x / (1.0f + __expf(-x)); }
__device__ __forceinline__ float gelu_t(float x) {
  float x3 = x * x * x;
  return 0.5f * x * (1.0f + tanhf(0.79788456080286535588f * (x + 0.044715f * x3)));
}
// A-fragment K-local index (ISA 7.12.2): e=0..7 -> hi*8+e ; e=8..15 -> 16+hi*8+(e-8)
__device__ __forceinline__ int aKl(int e, int hi) {
  int v = e >> 1;
  return ((v >= 4) ? 16 : 0) + (hi ? 8 : 0) + ((v & 3) << 1) + (e & 1);
}

#define WMMA_BF16(A, B, C) \
  __builtin_amdgcn_wmma_f32_16x16x32_bf16(false, (A), false, (B), (short)0, (C), false, false)

// ---------------------------------------------------------------------------
// bf16 packing kernels (conversion hoisted out of all hot loops)
// ---------------------------------------------------------------------------
__global__ void pack_bf16(const float* __restrict__ src, ushort_t* __restrict__ dst, long n) {
  long i = (long)blockIdx.x * blockDim.x + threadIdx.x;
  if (i < n) dst[i] = f2bfu(src[i]);
}
// W[K][N] f32 -> Wt[N][K] bf16 (write-coalesced)
__global__ void pack_transpose(const float* __restrict__ W, ushort_t* __restrict__ Wt,
                               int K, int N, long n) {
  long i = (long)blockIdx.x * blockDim.x + threadIdx.x;
  if (i >= n) return;
  int k = (int)(i % K);
  long nn = i / K;
  Wt[i] = f2bfu(W[(long)k * N + nn]);
}
// v (B,S,NH*128) bf16 -> vT (B,NH,128,S) bf16
__global__ void vtrans_pack(const ushort_t* __restrict__ vb, ushort_t* __restrict__ vT, long n) {
  long i = (long)blockIdx.x * blockDim.x + threadIdx.x;
  if (i >= n) return;
  int t = (int)(i & 1023);
  long r = i >> 10;
  int d = (int)(r & 127);
  int bh = (int)(r >> 7);
  int b = bh >> 2, h = bh & 3;
  vT[i] = vb[((long)(b * 1024 + t)) * 512 + h * 128 + d];
}

// ---------------------------------------------------------------------------
// Generic bf16 WMMA GEMM: C[M,N] = A[M,K] @ Wt[N][K]^T  (A,Wt pre-packed bf16)
// No LDS, no barriers: per K-step = 4 x b128 loads + 1 WMMA per wave.
// block = 128 threads (4 waves), block tile 16x64; blockIdx.z batches.
// ---------------------------------------------------------------------------
__global__ void __launch_bounds__(128)
gemm_bf16_wmma(const ushort_t* __restrict__ A, int lda,
               const ushort_t* __restrict__ Wt, int ldw,
               float* __restrict__ C, int ldc, int K,
               long sA, long sW, long sC) {
  const int wave = threadIdx.x >> 5;
  const int lane = threadIdx.x & 31;
  const int hi = lane >> 4;
  const int ln = lane & 15;
  const int m0 = blockIdx.y * 16;
  const int n  = (blockIdx.x * 4 + wave) * 16 + ln;
  A  += (long)blockIdx.z * sA;
  Wt += (long)blockIdx.z * sW;
  C  += (long)blockIdx.z * sC;
  const ushort_t* Ar = A + (long)(m0 + ln) * lda;
  const ushort_t* Wr = Wt + (long)n * ldw;
  v8f c;
#pragma unroll
  for (int r = 0; r < 8; ++r) c[r] = 0.0f;

  for (int k0 = 0; k0 < K; k0 += 32) {
    V16 a, b;
    a.u4[0] = *(const uint4*)(Ar + k0 + hi * 8);
    a.u4[1] = *(const uint4*)(Ar + k0 + 16 + hi * 8);
    b.u4[0] = *(const uint4*)(Wr + k0 + hi * 8);
    b.u4[1] = *(const uint4*)(Wr + k0 + 16 + hi * 8);
    if (k0 + 32 < K) __builtin_prefetch(Ar + k0 + 32, 0, 0);
    c = WMMA_BF16(a.v, b.v, c);
  }
#pragma unroll
  for (int r = 0; r < 8; ++r)
    C[(long)(m0 + r + hi * 8) * ldc + n] = c[r];   // C layout: M=r(+8), N=lane%16
}

// ---------------------------------------------------------------------------
// Embedding helpers
// ---------------------------------------------------------------------------
__global__ void renorm_rows(const float* __restrict__ src, float* __restrict__ dst, int cols) {
  __shared__ float s1[256];
  int row = blockIdx.x, tid = threadIdx.x;
  float v = (tid < cols) ? src[row * cols + tid] : 0.0f;
  s1[tid] = v * v;
  __syncthreads();
  for (int off = 128; off > 0; off >>= 1) {
    if (tid < off) s1[tid] += s1[tid + off];
    __syncthreads();
  }
  float nrm = sqrtf(s1[0]);
  float sc = fminf(1.0f, 2.0f / fmaxf(nrm, 1e-7f));
  if (tid < cols) dst[row * cols + tid] = v * sc;
}

__global__ void embed_kernel(const int* __restrict__ tok, const int* __restrict__ tis,
                             const float* __restrict__ seqn, const float* __restrict__ tisn,
                             float* __restrict__ x) {
  int row = blockIdx.x;        // b*1024+s
  int c = threadIdx.x;         // 0..255
  int b = row >> 10;
  int t = tok[row];
  float v = (c < 192) ? seqn[t * 192 + c] : tisn[tis[b] * 64 + (c - 192)];
  if (t == 0) v = 0.0f;
  x[(long)row * 256 + c] = v;
}

// ---------------------------------------------------------------------------
// LayerNorm over 256 channels; emits f32 AND packed bf16 (for WMMA consumers)
// ---------------------------------------------------------------------------
__global__ void layernorm256(const float* __restrict__ in, const float* __restrict__ w,
                             float* __restrict__ out, ushort_t* __restrict__ outB) {
  __shared__ float s1[256], s2[256];
  int row = blockIdx.x, tid = threadIdx.x;
  float v = in[(long)row * 256 + tid];
  s1[tid] = v; s2[tid] = v * v;
  __syncthreads();
  for (int off = 128; off > 0; off >>= 1) {
    if (tid < off) { s1[tid] += s1[tid + off]; s2[tid] += s2[tid + off]; }
    __syncthreads();
  }
  float mu = s1[0] * (1.0f / 256.0f);
  float var = s2[0] * (1.0f / 256.0f) - mu * mu;
  float r = (v - mu) * rsqrtf(var + 1e-5f) * w[tid];
  out[(long)row * 256 + tid] = r;
  outB[(long)row * 256 + tid] = f2bfu(r);
}

// ---------------------------------------------------------------------------
// Multi-head norm over DH (64/128); weight w[NH][DH], in-place on f32
// mode 0: rows = (b,s,h) head = row % 4 ; mode 1: rows = (b,h,s) head = (row/S)%4
// ---------------------------------------------------------------------------
__global__ void mhnorm_kernel(float* __restrict__ x, const float* __restrict__ w,
                              int dh, int mode, int S_) {
  __shared__ float s1[128], s2[128];
  int row = blockIdx.x, tid = threadIdx.x;
  int head = (mode == 0) ? (row & 3) : ((row / S_) & 3);
  float v = x[(long)row * dh + tid];
  s1[tid] = v; s2[tid] = v * v;
  __syncthreads();
  for (int off = blockDim.x >> 1; off > 0; off >>= 1) {
    if (tid < off) { s1[tid] += s1[tid + off]; s2[tid] += s2[tid + off]; }
    __syncthreads();
  }
  float inv = 1.0f / (float)dh;
  float mu = s1[0] * inv;
  float var = s2[0] * inv - mu * mu;
  x[(long)row * dh + tid] = (v - mu) * rsqrtf(var + 1e-5f) * w[head * dh + tid];
}

// ---------------------------------------------------------------------------
// Causal conv (K=4) + SiLU; f32-out (mLSTM) and bf16-out (sLSTM) variants
// ---------------------------------------------------------------------------
__device__ __forceinline__ float conv_val(const float* in, int inStride, int C,
                                          const float* w, const float* bconv, long i) {
  int c = (int)(i % C);
  long row = i / C;
  int s = (int)(row & 1023);
  float acc = bconv[c];
#pragma unroll
  for (int kk = 0; kk < 4; ++kk) {
    int ss = s + kk - 3;
    if (ss >= 0) acc += w[kk * C + c] * in[(row + kk - 3) * (long)inStride + c];
  }
  return siluf(acc);
}
__global__ void conv_silu_f32(const float* __restrict__ in, int inStride, int C,
                              const float* __restrict__ w, const float* __restrict__ bconv,
                              float* __restrict__ out, long n) {
  long i = (long)blockIdx.x * blockDim.x + threadIdx.x;
  if (i < n) out[i] = conv_val(in, inStride, C, w, bconv, i);
}
__global__ void conv_silu_bf16(const float* __restrict__ in, int inStride, int C,
                               const float* __restrict__ w, const float* __restrict__ bconv,
                               ushort_t* __restrict__ out, long n) {
  long i = (long)blockIdx.x * blockDim.x + threadIdx.x;
  if (i < n) out[i] = f2bfu(conv_val(in, inStride, C, w, bconv, i));
}

// ---------------------------------------------------------------------------
// 4x4 block-diagonal projection (mLSTM q/k/v), C = 512 ; emits packed bf16
// ---------------------------------------------------------------------------
__global__ void blockdiag4(const float* __restrict__ in, int inStride,
                           const float* __restrict__ w, ushort_t* __restrict__ out, long n) {
  long i = (long)blockIdx.x * blockDim.x + threadIdx.x;
  if (i >= n) return;
  int col = (int)(i % 512);
  long row = i / 512;
  int blk = col >> 2, o = col & 3;
  const float* ip = in + row * (long)inStride + blk * 4;
  const float* wp = w + blk * 16 + o * 4;
  out[i] = f2bfu(ip[0] * wp[0] + ip[1] * wp[1] + ip[2] * wp[2] + ip[3] * wp[3]);
}

// ---------------------------------------------------------------------------
// mLSTM i/f gate projections: gin = [q,k,v] (1536, bf16) @ (1536x4) + b
// outputs layout (B, NH, S)
// ---------------------------------------------------------------------------
__global__ void gates_mlstm(const ushort_t* __restrict__ q, const ushort_t* __restrict__ k,
                            const ushort_t* __restrict__ v,
                            const float* __restrict__ igw, const float* __restrict__ igb,
                            const float* __restrict__ fgw, const float* __restrict__ fgb,
                            float* __restrict__ ig_t, float* __restrict__ fg_t) {
  __shared__ float red[8][256];
  int row = blockIdx.x;            // b*1024+s
  int tid = threadIdx.x;
  int b = row >> 10, s = row & 1023;
  float aI[4] = {0, 0, 0, 0}, aF[4] = {0, 0, 0, 0};
  for (int kk = tid; kk < 1536; kk += 256) {
    float g;
    if (kk < 512)       g = bf2f(q[(long)row * 512 + kk]);
    else if (kk < 1024) g = bf2f(k[(long)row * 512 + kk - 512]);
    else                g = bf2f(v[(long)row * 512 + kk - 1024]);
#pragma unroll
    for (int o = 0; o < 4; ++o) {
      aI[o] += g * igw[kk * 4 + o];
      aF[o] += g * fgw[kk * 4 + o];
    }
  }
#pragma unroll
  for (int o = 0; o < 4; ++o) { red[o][tid] = aI[o]; red[4 + o][tid] = aF[o]; }
  __syncthreads();
  for (int off = 128; off > 0; off >>= 1) {
    if (tid < off) {
#pragma unroll
      for (int o = 0; o < 8; ++o) red[o][tid] += red[o][tid + off];
    }
    __syncthreads();
  }
  if (tid < 4)       ig_t[(b * 4 + tid) * 1024 + s] = red[tid][0] + igb[tid];
  else if (tid < 8)  { int o = tid - 4; fg_t[(b * 4 + o) * 1024 + s] = red[tid][0] + fgb[o]; }
}

// ---------------------------------------------------------------------------
// Per-(b,h) decay prep: lfc = cumsum(logsigmoid(fg)); a = ig - lfc; pm = prefmax(a)
// ---------------------------------------------------------------------------
__global__ void __launch_bounds__(1024)
decay_prep(const float* __restrict__ ig_t, const float* __restrict__ fg_t,
           float* __restrict__ lfc, float* __restrict__ aa, float* __restrict__ pm) {
  __shared__ float sb[1024];
  int bh = blockIdx.x, t = threadIdx.x;
  float lf = lsig(fg_t[bh * 1024 + t]);
  sb[t] = lf;
  __syncthreads();
  for (int off = 1; off < 1024; off <<= 1) {
    float u = (t >= off) ? sb[t - off] : 0.0f;
    __syncthreads();
    sb[t] += u;
    __syncthreads();
  }
  float lc = sb[t];
  lfc[bh * 1024 + t] = lc;
  float a = ig_t[bh * 1024 + t] - lc;
  aa[bh * 1024 + t] = a;
  __syncthreads();
  sb[t] = a;
  __syncthreads();
  for (int off = 1; off < 1024; off <<= 1) {
    float u = (t >= off) ? sb[t - off] : -3.0e38f;
    __syncthreads();
    sb[t] = fmaxf(sb[t], u);
    __syncthreads();
  }
  pm[bh * 1024 + t] = sb[t];
}

// ---------------------------------------------------------------------------
// Fused mLSTM attention; q/k bf16 (B,S,512), vT bf16 (B,NH,128,S); out f32.
// All WMMA fragments are pairs of b128 loads; P staged via 2KB LDS.
// ---------------------------------------------------------------------------
__global__ void __launch_bounds__(32)
attn_kernel(const ushort_t* __restrict__ q, const ushort_t* __restrict__ kbuf,
            const ushort_t* __restrict__ vT, const float* __restrict__ lfc,
            const float* __restrict__ aa, const float* __restrict__ pm,
            float* __restrict__ o) {
  __shared__ float pbuf[16][34];
  const int stile = blockIdx.x;
  const int bh = blockIdx.y;
  const int b = bh >> 2, h = bh & 3;
  const int lane = threadIdx.x;
  const int hi = lane >> 4, ln = lane & 15;
  const float scale = 0.0883883476483184f;  // 128^-0.5
  const long base = ((long)b * 1024) * 512 + h * 128;

  V16 qf[4];
  {
    const ushort_t* Qr = q + base + (long)(stile * 16 + ln) * 512;
#pragma unroll
    for (int f = 0; f < 4; ++f) {
      qf[f].u4[0] = *(const uint4*)(Qr + f * 32 + hi * 8);
      qf[f].u4[1] = *(const uint4*)(Qr + f * 32 + 16 + hi * 8);
    }
  }
  float pm8[8], lf8[8];
#pragma unroll
  for (int r = 0; r < 8; ++r) {
    int s = stile * 16 + r + hi * 8;
    pm8[r] = pm[bh * 1024 + s];
    lf8[r] = lfc[bh * 1024 + s];
  }
  v8f acc[8];
#pragma unroll
  for (int i = 0; i < 8; ++i)
#pragma unroll
    for (int r = 0; r < 8; ++r) acc[i][r] = 0.0f;
  float rs[8] = {0, 0, 0, 0, 0, 0, 0, 0};

  for (int tp = 0; tp <= stile / 2; ++tp) {
#pragma unroll
    for (int half = 0; half < 2; ++half) {
      int tt = 2 * tp + half;
      if (tt <= stile) {
        int t = tt * 16 + ln;
        const ushort_t* Kr = kbuf + base + (long)t * 512;
        v8f c;
#pragma unroll
        for (int r = 0; r < 8; ++r) c[r] = 0.0f;
#pragma unroll
        for (int f = 0; f < 4; ++f) {
          V16 kf;
          kf.u4[0] = *(const uint4*)(Kr + f * 32 + hi * 8);
          kf.u4[1] = *(const uint4*)(Kr + f * 32 + 16 + hi * 8);
          c = WMMA_BF16(qf[f].v, kf.v, c);
        }
        float at = aa[bh * 1024 + t];
#pragma unroll
        for (int r = 0; r < 8; ++r) {
          int s = stile * 16 + r + hi * 8;
          float p = (t <= s) ? c[r] * scale * __expf(at - pm8[r]) : 0.0f;
          rs[r] += p;
          pbuf[r + hi * 8][ln + half * 17] = p;
        }
      } else {
#pragma unroll
        for (int r = 0; r < 8; ++r) pbuf[r + hi * 8][ln + half * 17] = 0.0f;
      }
    }
    __syncthreads();
    v16bf pa;
#pragma unroll
    for (int e = 0; e < 16; ++e) {
      int kl = aKl(e, hi);
      pa[e] = bfh(f2bfu(pbuf[ln][(kl & 15) + (kl >> 4) * 17]));
    }
#pragma unroll
    for (int nt = 0; nt < 8; ++nt) {
      const ushort_t* Vr = vT + ((long)bh * 128 + nt * 16 + ln) * 1024;
      V16 vf;
      vf.u4[0] = *(const uint4*)(Vr + tp * 32 + hi * 8);
      vf.u4[1] = *(const uint4*)(Vr + tp * 32 + 16 + hi * 8);
      acc[nt] = WMMA_BF16(pa, vf.v, acc[nt]);
    }
    __syncthreads();
  }
#pragma unroll
  for (int r = 0; r < 8; ++r) {
    float v = rs[r];
    v += __shfl_xor(v, 1, 32);
    v += __shfl_xor(v, 2, 32);
    v += __shfl_xor(v, 4, 32);
    v += __shfl_xor(v, 8, 32);
    rs[r] = v;
  }
#pragma unroll
  for (int r = 0; r < 8; ++r) {
    int s = stile * 16 + r + hi * 8;
    float norm = fmaxf(fabsf(rs[r]), __expf(-(lf8[r] + pm8[r]))) + 1e-6f;
    float inv = 1.0f / norm;
#pragma unroll
    for (int nt = 0; nt < 8; ++nt)
      o[base + (long)s * 512 + nt * 16 + ln] = acc[nt][r] * inv;
  }
}

// ---------------------------------------------------------------------------
// Pointwise kernels
// ---------------------------------------------------------------------------
__global__ void add_inplace(float* __restrict__ dst, const float* __restrict__ src, long n) {
  long i = (long)blockIdx.x * blockDim.x + threadIdx.x;
  if (i < n) dst[i] += src[i];
}
// h_out(bf16) = (h + skip*xc) * silu(z)
__global__ void mlstm_post(const float* __restrict__ h, const float* __restrict__ xc,
                           const float* __restrict__ big, const float* __restrict__ skip,
                           ushort_t* __restrict__ out, long n) {
  long i = (long)blockIdx.x * blockDim.x + threadIdx.x;
  if (i >= n) return;
  int c = (int)(i % 512);
  long row = i / 512;
  float z = big[row * 1024 + 512 + c];
  out[i] = f2bfu((h[i] + skip[c] * xc[i]) * siluf(z));
}
__global__ void ffn_gate(const float* __restrict__ big, ushort_t* __restrict__ g, long n) {
  long i = (long)blockIdx.x * blockDim.x + threadIdx.x;
  if (i >= n) return;
  int c = (int)(i % 384);
  long row = i / 384;
  g[i] = f2bfu(gelu_t(big[row * 768 + c]) * big[row * 768 + 384 + c]);
}
__global__ void ytrans_add(float* __restrict__ x, const float* __restrict__ hb, long n) {
  long i = (long)blockIdx.x * blockDim.x + threadIdx.x;
  if (i >= n) return;
  int c = (int)(i % 256);
  long row = i / 256;
  int b = (int)(row >> 10), s = (int)(row & 1023);
  int h = c >> 6, d = c & 63;
  x[i] += hb[(((long)b * 4 + h) * 1024 + s) * 64 + d];
}

// ---------------------------------------------------------------------------
// sLSTM recurrence: single workgroup, 16 waves = (head,gate); R bf16 fragments
// preloaded with b128 loads; h state in LDS; 8 WMMAs/step.
// ---------------------------------------------------------------------------
__global__ void __launch_bounds__(512)
slstm_scan(const float* __restrict__ gates, const float* __restrict__ bias,
           const ushort_t* __restrict__ Rb, float* __restrict__ hbuf) {
  __shared__ float h_lds[1024];
  __shared__ float raw_lds[4][1024];
  const int tid = threadIdx.x;
  const int wave = tid >> 5;
  const int lane = tid & 31;
  const int hi = lane >> 4, ln = lane & 15;
  const int hh = wave >> 2, gg = wave & 3;
  const long GSTRIDE = 1048576;

  V16 rf[2][4];
  const ushort_t* Rp = Rb + (long)(hh * 4 + gg) * 4096;
#pragma unroll
  for (int kt = 0; kt < 2; ++kt)
#pragma unroll
    for (int nt = 0; nt < 4; ++nt) {
      const ushort_t* Rr = Rp + (nt * 16 + ln) * 64;
      rf[kt][nt].u4[0] = *(const uint4*)(Rr + kt * 32 + hi * 8);
      rf[kt][nt].u4[1] = *(const uint4*)(Rr + kt * 32 + 16 + hi * 8);
    }
  float cst[2] = {0, 0}, nst[2] = {0, 0}, mst[2] = {0, 0};
  h_lds[tid] = 0.0f;
  h_lds[tid + 512] = 0.0f;
  __syncthreads();

  for (int t = 0; t < 1024; ++t) {
    v16bf a2[2];
#pragma unroll
    for (int kt = 0; kt < 2; ++kt)
#pragma unroll
      for (int e = 0; e < 16; ++e) {
        int d = kt * 32 + aKl(e, hi);
        float hv = (ln < 4) ? h_lds[ln * 256 + hh * 64 + d] : 0.0f;
        a2[kt][e] = bfh(f2bfu(hv));
      }
#pragma unroll
    for (int nt = 0; nt < 4; ++nt) {
      v8f c;
#pragma unroll
      for (int r = 0; r < 8; ++r) c[r] = 0.0f;
      c = WMMA_BF16(a2[0], rf[0][nt].v, c);
      c = WMMA_BF16(a2[1], rf[1][nt].v, c);
      if (hi == 0) {
#pragma unroll
        for (int r = 0; r < 4; ++r)
          raw_lds[gg][r * 256 + hh * 64 + nt * 16 + ln] = c[r];
      }
    }
    __syncthreads();
#pragma unroll
    for (int j = 0; j < 2; ++j) {
      int idx = tid + j * 512;
      int b = idx >> 8;
      int rem = idx & 255;
      int h = rem >> 6, d = rem & 63;
      long grow = ((long)b * 1024 + t) * 256 + rem;
      float ir  = raw_lds[0][idx] + gates[0 * GSTRIDE + grow] + bias[(0 * 4 + h) * 64 + d];
      float fr  = raw_lds[1][idx] + gates[1 * GSTRIDE + grow] + bias[(1 * 4 + h) * 64 + d];
      float zr  = raw_lds[2][idx] + gates[2 * GSTRIDE + grow] + bias[(2 * 4 + h) * 64 + d];
      float orr = raw_lds[3][idx] + gates[3 * GSTRIDE + grow] + bias[(3 * 4 + h) * 64 + d];
      float lfm = mst[j] + lsig(fr);
      float mnew = fmaxf(ir, lfm);
      float igv = __expf(ir - mnew);
      float fgv = __expf(lfm - mnew);
      float cnew = fgv * cst[j] + igv * tanhf(zr);
      float nnew = fgv * nst[j] + igv;
      float hnew = (1.0f / (1.0f + __expf(-orr))) * cnew / nnew;
      cst[j] = cnew; nst[j] = nnew; mst[j] = mnew;
      h_lds[idx] = hnew;
      hbuf[(((long)b * 4 + h) * 1024 + t) * 64 + d] = hnew;
    }
    __syncthreads();
  }
}

// ---------------------------------------------------------------------------
// Final: LN(post_norm) on the 4 selected rows + MLP head -> out[b]
// ---------------------------------------------------------------------------
__global__ void head_kernel(const float* __restrict__ x, const float* __restrict__ pw,
                            const int* __restrict__ slen,
                            const float* __restrict__ w1, const float* __restrict__ b1,
                            const float* __restrict__ w2, const float* __restrict__ b2,
                            float* __restrict__ out) {
  __shared__ float xr[256], s1[256], s2[256];
  int b = blockIdx.x, tid = threadIdx.x;
  int idx = slen[b] - 1;
  idx = max(0, min(1023, idx));
  float v = x[((long)b * 1024 + idx) * 256 + tid];
  s1[tid] = v; s2[tid] = v * v;
  __syncthreads();
  for (int off = 128; off > 0; off >>= 1) {
    if (tid < off) { s1[tid] += s1[tid + off]; s2[tid] += s2[tid + off]; }
    __syncthreads();
  }
  float mu = s1[0] * (1.0f / 256.0f);
  float var = s2[0] * (1.0f / 256.0f) - mu * mu;
  xr[tid] = (v - mu) * rsqrtf(var + 1e-5f) * pw[tid];
  __syncthreads();
  float hv = 0.0f;
  if (tid < 128) {
    float acc = b1[tid];
    for (int k = 0; k < 256; ++k) acc += xr[k] * w1[k * 128 + tid];
    hv = fmaxf(acc, 0.0f) * w2[tid];
  }
  s1[tid] = (tid < 128) ? hv : 0.0f;
  __syncthreads();
  for (int off = 128; off > 0; off >>= 1) {
    if (tid < off) s1[tid] += s1[tid + off];
    __syncthreads();
  }
  if (tid == 0) out[b] = s1[0] + b2[0];
}

// ===========================================================================
// Host orchestration
// ===========================================================================
extern "C" void kernel_launch(void* const* d_in, const int* in_sizes, int n_in,
                              void* d_out, int out_size, void* d_ws, size_t ws_size,
                              hipStream_t stream) {
  (void)in_sizes; (void)n_in; (void)out_size; (void)ws_size;
  const int* tok  = (const int*)d_in[0];
  const int* tis  = (const int*)d_in[1];
  const int* slen = (const int*)d_in[2];
  // params: jax tree order — blocks[0..3] (dict keys sorted), then tail keys sorted
  auto P = [&](int i) { return (const float*)d_in[i]; };
  // mLSTM (sorted): conv_b,conv_w,down_w,fg_b,fg_w,ig_b,ig_w,k_w,ln_w,outnorm_w,q_w,skip,up_w,v_w
  // sLSTM (sorted, 'R'<lowercase): R,bias,conv_b,conv_w,ffn_down_w,ffn_up_w,ln1_w,ln2_w,outnorm_w,wf,wi,wo,wz
  const int MB[3] = {3, 30, 44};
  const int SB = 17;
  const int I_POSTW = 58, I_PB1 = 59, I_PB2 = 60, I_PW1 = 61, I_PW2 = 62,
            I_SEQ = 63, I_TIS = 64;

  float* pool = (float*)d_ws;
  const size_t M1 = 1u << 20;
  float* x   = pool + 0 * M1;        // (4096,256) f32
  float* xn  = pool + 1 * M1;        // (4096,256) f32
  float* big = pool + 2 * M1;        // up to (4096,1024) f32
  float* xc  = pool + 6 * M1;        // (4096,512) f32 (mLSTM conv out)
  float* sg  = pool + 8 * M1;        // sLSTM gates: 4 x 1M f32 (8M..12M)
  float* hb  = pool + 12 * M1;       // (4096,512) f32 attn out / slstm hbuf
  float* tmp = pool + 14 * M1;       // (4096,256) f32
  float* sm  = pool + 15 * M1;       // small scalars
  float* IGT = sm, *FGT = sm + 16384, *LFC = sm + 32768, *AA = sm + 49152,
       * PM = sm + 65536, *SEQN = sm + 81920, *TISN = sm + 94400;
  // bf16 arena (halves)
  ushort_t* bhp = (ushort_t*)(pool + 16 * M1);
  ushort_t* QB = bhp + 0 * M1;       // (4096,512) bf16
  ushort_t* KB = bhp + 2 * M1;
  ushort_t* VB = bhp + 4 * M1;
  ushort_t* VT = bhp + 6 * M1;       // (B,NH,128,S)
  ushort_t* A0 = bhp + 8 * M1;       // GEMM A buf (hb/ffn-gate/xc-slstm), 2M halves
  ushort_t* A1 = bhp + 10 * M1;      // xn bf16, 1M halves
  ushort_t* WT = bhp + 11 * M1;      // packed weights, 512K halves
  ushort_t* RB = bhp + 11 * M1 + 524288;  // packed R, 64K halves

  const long NTOK = 4096;
  auto NB = [](long n) { return (unsigned)((n + 255) / 256); };

  renorm_rows<<<65, 256, 0, stream>>>(P(I_SEQ), SEQN, 192);
  renorm_rows<<<30, 256, 0, stream>>>(P(I_TIS), TISN, 64);
  embed_kernel<<<NTOK, 256, 0, stream>>>(tok, tis, SEQN, TISN, x);

  auto run_mlstm = [&](int B0) {
    layernorm256<<<NTOK, 256, 0, stream>>>(x, P(B0 + 8), xn, A1);
    pack_transpose<<<NB(256 * 1024), 256, 0, stream>>>(P(B0 + 12), WT, 256, 1024, 256 * 1024);
    gemm_bf16_wmma<<<dim3(16, 256, 1), 128, 0, stream>>>(A1, 256, WT, 256,
                                                         big, 1024, 256, 0, 0, 0);
    long n5 = NTOK * 512;
    conv_silu_f32<<<NB(n5), 256, 0, stream>>>(big, 1024, 512, P(B0 + 1), P(B0 + 0), xc, n5);
    blockdiag4<<<NB(n5), 256, 0, stream>>>(xc, 512, P(B0 + 10), QB, n5);
    blockdiag4<<<NB(n5), 256, 0, stream>>>(xc, 512, P(B0 + 7), KB, n5);
    blockdiag4<<<NB(n5), 256, 0, stream>>>(big, 1024, P(B0 + 13), VB, n5);
    vtrans_pack<<<NB(n5), 256, 0, stream>>>(VB, VT, n5);
    gates_mlstm<<<NTOK, 256, 0, stream>>>(QB, KB, VB, P(B0 + 6), P(B0 + 5),
                                          P(B0 + 4), P(B0 + 3), IGT, FGT);
    decay_prep<<<16, 1024, 0, stream>>>(IGT, FGT, LFC, AA, PM);
    attn_kernel<<<dim3(64, 16, 1), 32, 0, stream>>>(QB, KB, VT, LFC, AA, PM, hb);
    mhnorm_kernel<<<16384, 128, 0, stream>>>(hb, P(B0 + 9), 128, 0, 1024);
    mlstm_post<<<NB(n5), 256, 0, stream>>>(hb, xc, big, P(B0 + 11), A0, n5);
    pack_transpose<<<NB(512 * 256), 256, 0, stream>>>(P(B0 + 2), WT, 512, 256, 512 * 256);
    gemm_bf16_wmma<<<dim3(4, 256, 1), 128, 0, stream>>>(A0, 512, WT, 512,
                                                        tmp, 256, 512, 0, 0, 0);
    add_inplace<<<NB(NTOK * 256), 256, 0, stream>>>(x, tmp, NTOK * 256);
  };

  auto run_slstm = [&](int B0) {
    layernorm256<<<NTOK, 256, 0, stream>>>(x, P(B0 + 6), xn, A1);
    long n2 = NTOK * 256;
    conv_silu_bf16<<<NB(n2), 256, 0, stream>>>(xn, 256, 256, P(B0 + 3), P(B0 + 2), A0, n2);
    // pack per-head gate weights (already N-major (o,i)) : wi,wf,wz,wo
    pack_bf16<<<NB(16384), 256, 0, stream>>>(P(B0 + 10), WT + 0 * 16384, 16384);
    pack_bf16<<<NB(16384), 256, 0, stream>>>(P(B0 + 9),  WT + 1 * 16384, 16384);
    pack_bf16<<<NB(16384), 256, 0, stream>>>(P(B0 + 12), WT + 2 * 16384, 16384);
    pack_bf16<<<NB(16384), 256, 0, stream>>>(P(B0 + 11), WT + 3 * 16384, 16384);
    gemm_bf16_wmma<<<dim3(1, 256, 4), 128, 0, stream>>>(A0, 256, WT + 0 * 16384, 64,
                                                        sg + 0 * M1, 256, 64, 64, 4096, 64);
    gemm_bf16_wmma<<<dim3(1, 256, 4), 128, 0, stream>>>(A0, 256, WT + 1 * 16384, 64,
                                                        sg + 1 * M1, 256, 64, 64, 4096, 64);
    gemm_bf16_wmma<<<dim3(1, 256, 4), 128, 0, stream>>>(A1, 256, WT + 2 * 16384, 64,
                                                        sg + 2 * M1, 256, 64, 64, 4096, 64);
    gemm_bf16_wmma<<<dim3(1, 256, 4), 128, 0, stream>>>(A1, 256, WT + 3 * 16384, 64,
                                                        sg + 3 * M1, 256, 64, 64, 4096, 64);
    pack_bf16<<<NB(65536), 256, 0, stream>>>(P(B0 + 0), RB, 65536);
    slstm_scan<<<1, 512, 0, stream>>>(sg, P(B0 + 1), RB, hb);
    mhnorm_kernel<<<16384, 64, 0, stream>>>(hb, P(B0 + 8), 64, 1, 1024);
    ytrans_add<<<NB(NTOK * 256), 256, 0, stream>>>(x, hb, NTOK * 256);
    layernorm256<<<NTOK, 256, 0, stream>>>(x, P(B0 + 7), xn, A1);
    pack_transpose<<<NB(256 * 768), 256, 0, stream>>>(P(B0 + 5), WT, 256, 768, 256 * 768);
    gemm_bf16_wmma<<<dim3(12, 256, 1), 128, 0, stream>>>(A1, 256, WT, 256,
                                                         big, 768, 256, 0, 0, 0);
    long n3 = NTOK * 384;
    ffn_gate<<<NB(n3), 256, 0, stream>>>(big, A0, n3);
    pack_transpose<<<NB(384 * 256), 256, 0, stream>>>(P(B0 + 4), WT, 384, 256, 384 * 256);
    gemm_bf16_wmma<<<dim3(4, 256, 1), 128, 0, stream>>>(A0, 384, WT, 384,
                                                        tmp, 256, 384, 0, 0, 0);
    add_inplace<<<NB(NTOK * 256), 256, 0, stream>>>(x, tmp, NTOK * 256);
  };

  run_mlstm(MB[0]);   // block 0
  run_slstm(SB);      // block 1
  run_mlstm(MB[1]);   // block 2
  run_mlstm(MB[2]);   // block 3

  head_kernel<<<4, 256, 0, stream>>>(x, P(I_POSTW), slen, P(I_PW1), P(I_PB1),
                                     P(I_PW2), P(I_PB2), (float*)d_out);
}
